// MambaLMPara_66786741453381
// MI455X (gfx1250) — matrix operand order repaired and checked by reading, not compiled, not run on hardware
//
#include <hip/hip_runtime.h>

// Mamba LM forward for MI455X (gfx1250, wave32, WMMA).
// All GEMMs run on v_wmma_f32_16x16x32_bf16 with in-register f32->bf16
// conversion (net is HBM/L2-bound; bf16 WMMA compute is essentially free).

#define NLAY  4
#define DMODEL 512
#define EDIM  1024
#define NSTATE 16
#define DCONV 4
#define DTRANK 32
#define VOCAB 32000
#define BATCH 2
#define SEQL  1024
#define MROWS (BATCH*SEQL)   // 2048

typedef __attribute__((ext_vector_type(16))) __bf16    v16bf;
typedef __attribute__((ext_vector_type(8)))  float     v8f;
typedef __attribute__((ext_vector_type(8)))  unsigned  v8u;

// ---------------- helpers ----------------

// Two f32 -> packed bf16x2 in one dword: round-to-nearest (ties away) via
// +0x8000, then a single v_perm_b32 grabs both high halves.
__device__ __forceinline__ unsigned bfpair(float lo, float hi) {
  return __builtin_amdgcn_perm(__builtin_bit_cast(unsigned, hi) + 0x8000u,
                               __builtin_bit_cast(unsigned, lo) + 0x8000u,
                               0x07060302u);
}

__device__ __forceinline__ v16bf pack16(float4 a, float4 b, float4 c, float4 d) {
  v8u u;
  u[0] = bfpair(a.x, a.y); u[1] = bfpair(a.z, a.w);
  u[2] = bfpair(b.x, b.y); u[3] = bfpair(b.z, b.w);
  u[4] = bfpair(c.x, c.y); u[5] = bfpair(c.z, c.w);
  u[6] = bfpair(d.x, d.y); u[7] = bfpair(d.z, d.w);
  return __builtin_bit_cast(v16bf, u);
}

__device__ __forceinline__ float sigmoidf_(float x) { return 1.0f / (1.0f + __expf(-x)); }
__device__ __forceinline__ float softplusf_(float x) {
  return (x > 20.0f) ? x : log1pf(__expf(x));
}

// ---------------- WMMA GEMM:  C[M,N] = A[M,K] * W[N,K]^T (+ C) ----------------
// One wave (32 threads) per 16(M) x 32(N) strip; 2 WMMAs per K-chunk share the
// A fragment. Fragment index mapping used here (A/B in contiguous k order):
//   A (16-bit 16x32): lane l: M = l&15; elem i -> K = i + 16*(l>>4)
//   B (16-bit 32x16): lane l: N = l&15; elem i -> K = i + 16*(l>>4)
//   C (f32 16x16)   : vgpr r: M = r + 8*(l>>4); N = l&15
// The 12 b128 loads of a K-chunk are issued as ONE hand-written load clause
// (s_clause 0xB) with forced-distinct destination VGPRs and a single
// s_wait_loadcnt: all L2 round-trips of a chunk overlap, and the working set
// (48 f32 regs + 16 acc regs) stays under the VGPR budget so the compiler's
// cross-iteration pipelining does not spill to scratch.
__global__ void gemm_wmma_bf16(const float* __restrict__ A, int lda,
                               const float* __restrict__ W, int ldw,
                               float* __restrict__ C, int ldc,
                               int K, int addC) {
  const int lane = threadIdx.x;          // block = exactly one wave32
  const int half = lane >> 4;
  const int l16  = lane & 15;
  const int tileM = blockIdx.x * 16;
  const int tileN = blockIdx.y * 32;

  const float* arow  = A + (size_t)(tileM + l16) * lda;
  const float* wrow0 = W + (size_t)(tileN +  0 + l16) * ldw;
  const float* wrow1 = W + (size_t)(tileN + 16 + l16) * ldw;

  v8f c0 = {}; v8f c1 = {};

  for (int k0 = 0; k0 < K; k0 += 32) {
    __builtin_prefetch(arow + k0 + 128, 0, 1);   // global_prefetch_b8 ahead

    // Base addresses for this chunk (immediate offsets cover the rest):
    //   A: pa +0, +16, +32, +48 bytes ; each B row: +0, +16, +32, +48 bytes
    const float* pa = arow  + k0 + 16 * half;    // A2/A3 live at +32B/+48B
    const float* p0 = wrow0 + k0 + 16 * half;
    const float* p1 = wrow1 + k0 + 16 * half;

    float4 A0, A1, A2, A3;
    float4 B00, B01, B02, B03, B10, B11, B12, B13;

    asm volatile(
        "s_clause 0xb\n\t"                                // 12-load clause
        "global_load_b128 %0,  %12, off\n\t"
        "global_load_b128 %1,  %12, off offset:16\n\t"
        "global_load_b128 %2,  %12, off offset:32\n\t"
        "global_load_b128 %3,  %12, off offset:48\n\t"
        "global_load_b128 %4,  %13, off\n\t"
        "global_load_b128 %5,  %13, off offset:16\n\t"
        "global_load_b128 %6,  %13, off offset:32\n\t"
        "global_load_b128 %7,  %13, off offset:48\n\t"
        "global_load_b128 %8,  %14, off\n\t"
        "global_load_b128 %9,  %14, off offset:16\n\t"
        "global_load_b128 %10, %14, off offset:32\n\t"
        "global_load_b128 %11, %14, off offset:48\n\t"
        "s_wait_loadcnt 0x0"
        : "=v"(A0),  "=v"(A1),  "=v"(A2),  "=v"(A3),
          "=v"(B00), "=v"(B01), "=v"(B02), "=v"(B03),
          "=v"(B10), "=v"(B11), "=v"(B12), "=v"(B13)
        : "v"(pa), "v"(p0), "v"(p1));

    // ---- convert to bf16 fragments ----
    const v16bf af  = pack16(A0,  A1,  A2,  A3);
    const v16bf bf0 = pack16(B00, B01, B02, B03);
    const v16bf bf1 = pack16(B10, B11, B12, B13);

    // ---- 2 WMMAs sharing the A fragment ----
    c0 = __builtin_amdgcn_wmma_f32_16x16x32_bf16(false, af, false, bf0,
                                                 (short)0, c0, false, false);
    c1 = __builtin_amdgcn_wmma_f32_16x16x32_bf16(false, af, false, bf1,
                                                 (short)0, c1, false, false);
  }

  // Epilogue: hoist the uniform addC test so each path is a straight-line
  // unrolled store loop (no per-element uniform branching).
  float* crow0 = C + (size_t)(tileM + 8 * half) * ldc + tileN + l16;
  float* crow1 = crow0 + 16;
  if (addC) {
#pragma unroll
    for (int r = 0; r < 8; ++r) {
      float* q0 = crow0 + (size_t)r * ldc;
      float* q1 = crow1 + (size_t)r * ldc;
      *q0 += c0[r];
      *q1 += c1[r];
    }
  } else {
#pragma unroll
    for (int r = 0; r < 8; ++r) {
      crow0[(size_t)r * ldc] = c0[r];
      crow1[(size_t)r * ldc] = c1[r];
    }
  }
}

// ---------------- RMSNorm (block per row) ----------------
__global__ void rmsnorm_kernel(const float* __restrict__ x,
                               const float* __restrict__ w,
                               float* __restrict__ out, int cols) {
  const int row = blockIdx.x;
  const float* xr = x + (size_t)row * cols;
  float* orow = out + (size_t)row * cols;
  float ss = 0.f;
  for (int c = threadIdx.x; c < cols; c += blockDim.x) { float v = xr[c]; ss += v * v; }
  __shared__ float red[256];
  red[threadIdx.x] = ss;
  __syncthreads();
  for (int s = blockDim.x >> 1; s > 0; s >>= 1) {
    if ((int)threadIdx.x < s) red[threadIdx.x] += red[threadIdx.x + s];
    __syncthreads();
  }
  const float scale = rsqrtf(red[0] / (float)cols + 1e-5f);
  for (int c = threadIdx.x; c < cols; c += blockDim.x) orow[c] = xr[c] * scale * w[c];
}

// ---------------- causal depthwise conv (DC=4) + bias + SiLU ----------------
// xi lives in xz[..., 0:ED] (row stride 2*ED); output compact (row stride ED)
__global__ void conv_silu_kernel(const float* __restrict__ xz,
                                 const float* __restrict__ cw,
                                 const float* __restrict__ cb,
                                 float* __restrict__ out) {
  const size_t idx = (size_t)blockIdx.x * blockDim.x + threadIdx.x;
  if (idx >= (size_t)MROWS * EDIM) return;
  const int e = (int)(idx % EDIM);
  const int t = (int)((idx / EDIM) % SEQL);
  const int b = (int)(idx / ((size_t)EDIM * SEQL));
  float acc = cb[e];
#pragma unroll
  for (int j = 0; j < DCONV; ++j) {
    const int ts = t - (DCONV - 1) + j;
    if (ts >= 0)
      acc += cw[e * DCONV + j] * xz[((size_t)(b * SEQL + ts) * 2 * EDIM) + e];
  }
  out[idx] = acc * sigmoidf_(acc);
}

// ---------------- selective scan: one LANE per (b, e, n) state ----------------
// 16 lanes form one channel group; per-step y = sum_n(s_n * C_n) is 4 shfl_xor
// butterflies (masks 1/2/4/8 stay inside the 16-lane group on wave32).
// 32768 threads (1024 waves) hide the 1024-step sequential-exp latency chain.
// Fuses softplus(dt)+bias, dA=exp, D skip, and SiLU(z) gating.
__global__ void scan_kernel(const float* __restrict__ dpre,   // (M, ED) pre-softplus
                            const float* __restrict__ xcv,    // (M, ED) conv+silu
                            const float* __restrict__ xz,     // (M, 2ED) for z
                            const float* __restrict__ dbc,    // (M, 64): [dt | B | C]
                            const float* __restrict__ A_log,  // (ED, N)
                            const float* __restrict__ dt_b,   // (ED)
                            const float* __restrict__ Dp,     // (ED)
                            float* __restrict__ y) {          // (M, ED)
  const int gid = blockIdx.x * blockDim.x + threadIdx.x;      // 0 .. B*ED*16-1
  const int n  = gid & (NSTATE - 1);
  const int ch = gid >> 4;                                    // 0 .. B*ED-1
  const int e = ch % EDIM;
  const int b = ch / EDIM;

  const float an  = -__expf(A_log[(size_t)e * NSTATE + n]);
  const float dtb = dt_b[e];
  const float dpe = Dp[e];
  float st = 0.f;

  for (int t = 0; t < SEQL; ++t) {
    const size_t m = (size_t)(b * SEQL + t);
    const float delta = softplusf_(dpre[m * EDIM + e] + dtb);  // wave-broadcast read
    const float xi = xcv[m * EDIM + e];
    const float* row = dbc + m * (DTRANK + 2 * NSTATE);
    st = __expf(delta * an) * st + (delta * xi) * row[DTRANK + n];   // B_n
    float part = st * row[DTRANK + NSTATE + n];                      // C_n
    part += __shfl_xor(part, 1, 32);
    part += __shfl_xor(part, 2, 32);
    part += __shfl_xor(part, 4, 32);
    part += __shfl_xor(part, 8, 32);
    if (n == 0) {
      const float z = xz[m * 2 * EDIM + EDIM + e];
      y[m * EDIM + e] = (part + dpe * xi) * (z * sigmoidf_(z));
    }
  }
}

// ---------------- host side ----------------

static void launch_gemm(const float* A, int lda, const float* W, int ldw,
                        float* C, int ldc, int M, int N, int K, int addC,
                        hipStream_t s) {
  dim3 grid(M / 16, N / 32);
  gemm_wmma_bf16<<<grid, dim3(32), 0, s>>>(A, lda, W, ldw, C, ldc, K, addC);
}

extern "C" void kernel_launch(void* const* d_in, const int* in_sizes, int n_in,
                              void* d_out, int out_size, void* d_ws, size_t ws_size,
                              hipStream_t stream) {
  (void)in_sizes; (void)n_in; (void)out_size; (void)ws_size;

  const float* x       = (const float*)d_in[0];   // (B, L, DM)
  const float* norm_w  = (const float*)d_in[1];   // (NL, DM)
  const float* in_w    = (const float*)d_in[2];   // (NL, 2ED, DM)
  const float* conv_w  = (const float*)d_in[3];   // (NL, ED, 1, DC)
  const float* conv_b  = (const float*)d_in[4];   // (NL, ED)
  const float* xproj_w = (const float*)d_in[5];   // (NL, DTR+2N, ED)
  const float* dt_w    = (const float*)d_in[6];   // (NL, ED, DTR)
  const float* dt_b    = (const float*)d_in[7];   // (NL, ED)
  const float* A_log   = (const float*)d_in[8];   // (NL, ED, N)
  const float* D_p     = (const float*)d_in[9];   // (NL, ED)
  const float* out_w   = (const float*)d_in[10];  // (NL, DM, ED)
  const float* normf_w = (const float*)d_in[11];  // (DM)
  const float* head_w  = (const float*)d_in[12];  // (V, DM)
  float* logits = (float*)d_out;                  // (B, L, V)

  // workspace carve-up (floats), all 16B-aligned sizes
  float* ws   = (float*)d_ws;
  float* xr   = ws;                    // residual stream  (M, DM)   1,048,576
  float* h    = xr   + (size_t)MROWS * DMODEL;      // normed       1,048,576
  float* xzb  = h    + (size_t)MROWS * DMODEL;      // (M, 2ED)     4,194,304
  float* xcv  = xzb  + (size_t)MROWS * 2 * EDIM;    // (M, ED)      2,097,152
  float* dbc  = xcv  + (size_t)MROWS * EDIM;        // (M, 64)        131,072
  float* dpre = dbc  + (size_t)MROWS * (DTRANK + 2 * NSTATE);  //    2,097,152
  float* ybuf = dpre + (size_t)MROWS * EDIM;        // (M, ED)      2,097,152

  // residual stream starts as the input (graph-capture-safe d2d copy)
  hipMemcpyAsync(xr, x, (size_t)MROWS * DMODEL * sizeof(float),
                 hipMemcpyDeviceToDevice, stream);

  const int ew_blocks = (MROWS * EDIM + 255) / 256;

  for (int i = 0; i < NLAY; ++i) {
    const float* nw  = norm_w  + (size_t)i * DMODEL;
    const float* iw  = in_w    + (size_t)i * 2 * EDIM * DMODEL;
    const float* cw  = conv_w  + (size_t)i * EDIM * DCONV;
    const float* cb  = conv_b  + (size_t)i * EDIM;
    const float* xw  = xproj_w + (size_t)i * (DTRANK + 2 * NSTATE) * EDIM;
    const float* dw  = dt_w    + (size_t)i * EDIM * DTRANK;
    const float* db  = dt_b    + (size_t)i * EDIM;
    const float* al  = A_log   + (size_t)i * EDIM * NSTATE;
    const float* dp  = D_p     + (size_t)i * EDIM;
    const float* ow  = out_w   + (size_t)i * DMODEL * EDIM;

    // 1) pre-norm
    rmsnorm_kernel<<<MROWS, 256, 0, stream>>>(xr, nw, h, DMODEL);
    // 2) in_proj: (M,2ED) = h(M,DM) * in_w(2ED,DM)^T
    launch_gemm(h, DMODEL, iw, DMODEL, xzb, 2 * EDIM, MROWS, 2 * EDIM, DMODEL, 0, stream);
    // 3) causal depthwise conv + bias + SiLU on xi
    conv_silu_kernel<<<ew_blocks, 256, 0, stream>>>(xzb, cw, cb, xcv);
    // 4) x_proj: (M,64) = xcv(M,ED) * xproj_w(64,ED)^T
    launch_gemm(xcv, EDIM, xw, EDIM, dbc, DTRANK + 2 * NSTATE,
                MROWS, DTRANK + 2 * NSTATE, EDIM, 0, stream);
    // 5) dt_proj: (M,ED) = dbc[:, :DTR](lda=64) * dt_w(ED,DTR)^T
    launch_gemm(dbc, DTRANK + 2 * NSTATE, dw, DTRANK, dpre, EDIM,
                MROWS, EDIM, DTRANK, 0, stream);
    // 6) selective scan + D skip + SiLU(z) gate (fused), lane-per-state
    scan_kernel<<<(BATCH * EDIM * NSTATE) / 256, 256, 0, stream>>>(
        dpre, xcv, xzb, dbc, al, db, dp, ybuf);
    // 7) out_proj with residual accumulate: xr += y * out_w^T
    launch_gemm(ybuf, EDIM, ow, EDIM, xr, DMODEL, MROWS, DMODEL, EDIM, 1, stream);
  }

  // final norm + LM head: logits(M,V) = h(M,DM) * head_w(V,DM)^T
  rmsnorm_kernel<<<MROWS, 256, 0, stream>>>(xr, normf_w, h, DMODEL);
  launch_gemm(h, DMODEL, head_w, DMODEL, logits, VOCAB, MROWS, VOCAB, DMODEL, 0, stream);
}
